// InteractionBlock_24172075942148
// MI455X (gfx1250) — compile-verified, matrix operand order checked by reading, not run on hardware
//
#include <hip/hip_runtime.h>
#include <hip/hip_bf16.h>
#include <stdint.h>

// ---------------------------------------------------------------------------
// Types for CDNA5 WMMA (wave32): 16x16x32 bf16 -> f32
// ---------------------------------------------------------------------------
typedef __attribute__((ext_vector_type(16))) __bf16          v16bf;
typedef __attribute__((ext_vector_type(8)))  __bf16          v8bf;
typedef __attribute__((ext_vector_type(8)))  float           v8f;
typedef __attribute__((ext_vector_type(8)))  unsigned short  ush8;

__device__ __forceinline__ unsigned short f2bf(float f) {
  unsigned u = __float_as_uint(f);
  u += 0x7FFFu + ((u >> 16) & 1u);          // round-to-nearest-even
  return (unsigned short)(u >> 16);
}
// monotone float <-> uint mapping for atomicMax-based segment max
__device__ __forceinline__ unsigned f_sort(float f) {
  unsigned u = __float_as_uint(f);
  return (u & 0x80000000u) ? ~u : (u | 0x80000000u);
}
__device__ __forceinline__ float f_unsort(unsigned s) {
  return __uint_as_float((s & 0x80000000u) ? (s ^ 0x80000000u) : ~s);
}

// ---------------------------------------------------------------------------
// Weight convert: W[K][N] fp32 -> Wt[N][K] bf16 (B operand, column-per-lane)
// ---------------------------------------------------------------------------
__global__ void wconv_kernel(const float* __restrict__ W, unsigned short* __restrict__ Wt,
                             int K, int N) {
  int i = blockIdx.x * blockDim.x + threadIdx.x;
  if (i >= K * N) return;
  int n = i / K, k = i - n * K;
  Wt[i] = f2bf(W[(size_t)k * N + n]);
}
// pack up to 4 bias segments into one contiguous vector
__global__ void pack_bias_kernel(float* __restrict__ dst,
                                 const float* b0, const float* b1,
                                 const float* b2, const float* b3,
                                 int seglen, int nseg) {
  int i = blockIdx.x * blockDim.x + threadIdx.x;
  if (i >= seglen * nseg) return;
  int s = i / seglen, o = i - s * seglen;
  const float* p = (s == 0) ? b0 : (s == 1) ? b1 : (s == 2) ? b2 : b3;
  dst[i] = p[o];
}

// ---------------------------------------------------------------------------
// GEMM: C[M][ldc] = act(A[M][K](f32->bf16) * Bt[N][K](bf16) + bias[N])
// Block: 256 thr = 8 waves; tile 64x64; wave -> one 16-row strip, two 16-col
// tiles (shared A fragment). LDS staged, padded stride 40 (80B). Next K-tile
// prefetched (global_prefetch_b8) while current tile runs through WMMA.
// ---------------------------------------------------------------------------
__global__ __launch_bounds__(256)
void gemm_bf16_kernel(const float* __restrict__ A, const unsigned short* __restrict__ Bt,
                      const float* __restrict__ bias, float* __restrict__ C,
                      int M, int N, int K, int relu, int ldc) {
  __shared__ unsigned short As[64 * 40];
  __shared__ unsigned short Bs[64 * 40];
  const int t    = threadIdx.x;
  const int lane = t & 31;
  const int wave = t >> 5;
  const int mt   = wave & 3;          // 16-row strip index
  const int nsel = wave >> 2;         // 0/1 -> n-tiles nsel*16, nsel*16+32
  const int m0   = blockIdx.x * 64;
  const int n0   = blockIdx.y * 64;
  const int half = lane >> 4;
  const int mr   = lane & 15;
  const int lrow = t >> 2;            // row of tile this thread stages
  const int kq   = (t & 3) * 8;       // 0,8,16,24

  v8f c0 = {}; v8f c1 = {};

  for (int k0 = 0; k0 < K; k0 += 32) {
    __syncthreads();
    {
      float4 f0 = make_float4(0.f, 0.f, 0.f, 0.f), f1 = f0;
      int rg = m0 + lrow;
      if (rg < M) {
        const float* p = A + (size_t)rg * K + (k0 + kq);
        f0 = *(const float4*)p;
        f1 = *(const float4*)(p + 4);
        if (k0 + 32 < K) __builtin_prefetch(p + 32, 0, 3);   // next A K-tile
      }
      ush8 ua;
      ua[0] = f2bf(f0.x); ua[1] = f2bf(f0.y); ua[2] = f2bf(f0.z); ua[3] = f2bf(f0.w);
      ua[4] = f2bf(f1.x); ua[5] = f2bf(f1.y); ua[6] = f2bf(f1.z); ua[7] = f2bf(f1.w);
      *(ush8*)&As[lrow * 40 + kq] = ua;

      ush8 ub = {};
      int ng = n0 + lrow;
      if (ng < N) {
        const unsigned short* pb = Bt + (size_t)ng * K + (k0 + kq);
        ub = *(const ush8*)pb;
        if (k0 + 32 < K) __builtin_prefetch(pb + 32, 0, 3);  // next B K-tile
      }
      *(ush8*)&Bs[lrow * 40 + kq] = ub;
    }
    __syncthreads();

    // A fragment: lanes 0-15 K{0..7,16..23}, lanes 16-31 K{8..15,24..31}
    const unsigned short* pa = &As[(mt * 16 + mr) * 40 + half * 8];
    v8bf alo = *(const v8bf*)pa;
    v8bf ahi = *(const v8bf*)(pa + 16);
    v16bf fa = __builtin_shufflevector(alo, ahi, 0,1,2,3,4,5,6,7,8,9,10,11,12,13,14,15);

    const unsigned short* pb0 = &Bs[(nsel * 16 + mr) * 40 + half * 8];
    v8bf b0l = *(const v8bf*)pb0;
    v8bf b0h = *(const v8bf*)(pb0 + 16);
    v16bf fb0 = __builtin_shufflevector(b0l, b0h, 0,1,2,3,4,5,6,7,8,9,10,11,12,13,14,15);

    const unsigned short* pb1 = &Bs[(nsel * 16 + 32 + mr) * 40 + half * 8];
    v8bf b1l = *(const v8bf*)pb1;
    v8bf b1h = *(const v8bf*)(pb1 + 16);
    v16bf fb1 = __builtin_shufflevector(b1l, b1h, 0,1,2,3,4,5,6,7,8,9,10,11,12,13,14,15);

    c0 = __builtin_amdgcn_wmma_f32_16x16x32_bf16(false, fa, false, fb0, (short)0, c0, false, false);
    c1 = __builtin_amdgcn_wmma_f32_16x16x32_bf16(false, fa, false, fb1, (short)0, c1, false, false);
  }

  const int col0 = n0 + nsel * 16 + mr;
  const int col1 = col0 + 32;
  const float bb0 = bias ? bias[col0] : 0.f;
  const float bb1 = bias ? bias[col1] : 0.f;
  const int rb = m0 + mt * 16 + half * 8;
#pragma unroll
  for (int r = 0; r < 8; ++r) {
    int row = rb + r;
    if (row < M) {
      float x0 = c0[r] + bb0;
      float x1 = c1[r] + bb1;
      if (relu) { x0 = fmaxf(x0, 0.f); x1 = fmaxf(x1, 0.f); }
      C[(size_t)row * ldc + col0] = x0;
      C[(size_t)row * ldc + col1] = x1;
    }
  }
}

// ---------------------------------------------------------------------------
// Elementwise / scatter kernels
// ---------------------------------------------------------------------------
__global__ void fill_u32_kernel(unsigned* p, unsigned v, long n) {
  long i = (long)blockIdx.x * blockDim.x + threadIdx.x;
  if (i < n) p[i] = v;
}
__global__ void gine_msg_kernel(const float* __restrict__ x, const float* __restrict__ eg,
                                const int* __restrict__ src, const int* __restrict__ dst,
                                float* __restrict__ agg, int E) {
  long i = (long)blockIdx.x * blockDim.x + threadIdx.x;
  if (i >= (long)E * 64) return;
  int e = (int)(i >> 6), f = (int)(i & 63);
  float m = x[(size_t)src[e] * 64 + f] + eg[i];
  m = fmaxf(m, 0.f);
  atomicAdd(&agg[(size_t)dst[e] * 64 + f], m);
}
__global__ void gine_comb_kernel(const float* __restrict__ x, const float* __restrict__ agg,
                                 const float* __restrict__ eps, float* __restrict__ out, long n) {
  long i = (long)blockIdx.x * blockDim.x + threadIdx.x;
  if (i < n) out[i] = (1.0f + eps[0]) * x[i] + agg[i];
}
// q/k live in the fused [n][ld] projection buffer
__global__ void tc_score_kernel(const float* __restrict__ q, const float* __restrict__ k,
                                const float* __restrict__ ep,
                                const int* __restrict__ src, const int* __restrict__ dst,
                                float* __restrict__ s_out, unsigned* __restrict__ mbuf,
                                int E, int ld) {
  long i = (long)blockIdx.x * blockDim.x + threadIdx.x;
  if (i >= (long)E * 4) return;
  int e = (int)(i >> 2), h = (int)(i & 3);
  const float4* qp = (const float4*)(q + (size_t)dst[e] * ld + h * 64);
  const float4* kp = (const float4*)(k + (size_t)src[e] * ld + h * 64);
  const float4* pp = (const float4*)(ep + (size_t)e * 256 + h * 64);
  float acc = 0.f;
#pragma unroll 4
  for (int j = 0; j < 16; ++j) {
    float4 a = qp[j], b = kp[j], c = pp[j];
    acc += a.x * (b.x + c.x) + a.y * (b.y + c.y) + a.z * (b.z + c.z) + a.w * (b.w + c.w);
  }
  acc *= 0.125f;   // 1/sqrt(64)
  s_out[i] = acc;
  atomicMax(&mbuf[(size_t)dst[e] * 4 + h], f_sort(acc));
}
__global__ void seg_exp_kernel(const float* __restrict__ s, const unsigned* __restrict__ mbuf,
                               const int* __restrict__ dst, float* __restrict__ ex,
                               float* __restrict__ den, int E, int H) {
  long i = (long)blockIdx.x * blockDim.x + threadIdx.x;
  if (i >= (long)E * H) return;
  int e = (int)(i / H), h = (int)(i - (long)e * H);
  float m = f_unsort(mbuf[(size_t)dst[e] * H + h]);
  float v = expf(s[i] - m);
  ex[i] = v;
  atomicAdd(&den[(size_t)dst[e] * H + h], v);
}
__global__ void tc_scatter_kernel(const float* __restrict__ v, int ldv,
                                  const float* __restrict__ ep,
                                  const float* __restrict__ ex, const float* __restrict__ den,
                                  const int* __restrict__ src, const int* __restrict__ dst,
                                  float* __restrict__ out, int E) {
  long i = (long)blockIdx.x * blockDim.x + threadIdx.x;
  if (i >= (long)E * 256) return;
  int e = (int)(i >> 8), c = (int)(i & 255), h = c >> 6;
  int d = dst[e];
  float a = ex[(size_t)e * 4 + h] / (den[(size_t)d * 4 + h] + 1e-16f);
  float val = a * (v[(size_t)src[e] * ldv + c] + ep[i]);
  atomicAdd(&out[(size_t)d * 256 + c], val);
}
// out[n][256] += skip[n][lds_] (skip = slice of fused projection buffer)
__global__ void vadd_strided_kernel(float* __restrict__ out, const float* __restrict__ skip,
                                    int lds_, long n) {
  long i = (long)blockIdx.x * blockDim.x + threadIdx.x;
  if (i >= n) return;
  long row = i >> 8; int c = (int)(i & 255);
  out[i] += skip[(size_t)row * lds_ + c];
}
__global__ void gat_score_kernel(const float* __restrict__ xl, const float* __restrict__ xr,
                                 int ld, const float* __restrict__ eg,
                                 const float* __restrict__ att,
                                 const int* __restrict__ src, const int* __restrict__ dst,
                                 float* __restrict__ s_out, unsigned* __restrict__ mbuf, int E) {
  int e = blockIdx.x * blockDim.x + threadIdx.x;
  if (e >= E) return;
  const float4* pl = (const float4*)(xl + (size_t)src[e] * ld);
  const float4* pr = (const float4*)(xr + (size_t)dst[e] * ld);
  const float4* pe = (const float4*)(eg + (size_t)e * 64);
  const float4* pa = (const float4*)att;
  float acc = 0.f;
#pragma unroll 4
  for (int j = 0; j < 16; ++j) {
    float4 a = pl[j], b = pr[j], c = pe[j], w = pa[j];
    float z;
    z = a.x + b.x + c.x; z = (z > 0.f) ? z : 0.2f * z; acc += w.x * z;
    z = a.y + b.y + c.y; z = (z > 0.f) ? z : 0.2f * z; acc += w.y * z;
    z = a.z + b.z + c.z; z = (z > 0.f) ? z : 0.2f * z; acc += w.z * z;
    z = a.w + b.w + c.w; z = (z > 0.f) ? z : 0.2f * z; acc += w.w * z;
  }
  s_out[e] = acc;
  atomicMax(&mbuf[dst[e]], f_sort(acc));
}
__global__ void gat_scatter_kernel(const float* __restrict__ xl, int ld,
                                   const float* __restrict__ ex, const float* __restrict__ den,
                                   const int* __restrict__ src, const int* __restrict__ dst,
                                   float* __restrict__ out, int E) {
  long i = (long)blockIdx.x * blockDim.x + threadIdx.x;
  if (i >= (long)E * 64) return;
  int e = (int)(i >> 6), f = (int)(i & 63);
  int d = dst[e];
  float a = ex[e] / (den[d] + 1e-16f);
  atomicAdd(&out[(size_t)d * 64 + f], a * xl[(size_t)src[e] * ld + f]);
}
__global__ void add_bias_kernel(float* __restrict__ out, const float* __restrict__ bo, long n) {
  long i = (long)blockIdx.x * blockDim.x + threadIdx.x;
  if (i < n) out[i] += bo[i & 63];
}
__global__ void gather_rows_kernel(const float* __restrict__ x, const int* __restrict__ idx,
                                   float* __restrict__ out, int E) {
  long i = (long)blockIdx.x * blockDim.x + threadIdx.x;
  if (i >= (long)E * 64) return;
  int e = (int)(i >> 6), f = (int)(i & 63);
  out[i] = x[(size_t)idx[e] * 64 + f];
}
__global__ void combine_x_kernel(const float* __restrict__ x, const float* __restrict__ x2,
                                 const float* __restrict__ w, float* __restrict__ o, long n) {
  long i = (long)blockIdx.x * blockDim.x + threadIdx.x;
  if (i < n) o[i] = (x[i] * w[0] + x2[i] * w[1] + x[i] * w[2]) * (1.0f / 3.0f);
}
__global__ void combine_ea_kernel(const float* __restrict__ a1, const float* __restrict__ ea,
                                  const float* __restrict__ a3, const float* __restrict__ w,
                                  float* __restrict__ o, long n) {
  long i = (long)blockIdx.x * blockDim.x + threadIdx.x;
  if (i < n) o[i] = (a1[i] * w[0] + ea[i] * w[1] + a3[i] * w[2]) * (1.0f / 3.0f);
}
__global__ void copy_i32_kernel(const int* __restrict__ in, int* __restrict__ out, long n) {
  long i = (long)blockIdx.x * blockDim.x + threadIdx.x;
  if (i < n) out[i] = in[i];
}

// ---------------------------------------------------------------------------
// Host-side orchestration
// ---------------------------------------------------------------------------
struct GineP { const float *We,*be,*W1,*b1,*W2,*b2,*eps; };
struct TcP   { const float *Wq,*bq,*Wk,*bk,*Wv,*bv,*We,*Ws,*bs; };
struct GatP  { const float *Wl,*bl,*Wr,*br,*We,*att,*bo; };
struct SeqP  { GineP g; TcP t1, t2; GatP ga; };

struct Scratch {
  float *FA, *FB, *FQKVS, *FEP, *FEG, *FAGG, *FES, *FEX, *FDEN, *BIASP;
  unsigned *UM;
  unsigned short *WBF;
};

static inline int blks(long n) { return (int)((n + 255) / 256); }

static void conv_w(const float* W, unsigned short* Wt, int K, int N, hipStream_t s) {
  wconv_kernel<<<blks((long)K * N), 256, 0, s>>>(W, Wt, K, N);
}
static void gemm(const float* A, const unsigned short* Bt, const float* bias, float* C,
                 int M, int N, int K, int relu, int ldc, hipStream_t s) {
  dim3 g((M + 63) / 64, N / 64);
  gemm_bf16_kernel<<<g, 256, 0, s>>>(A, Bt, bias, C, M, N, K, relu, ldc);
}
static void zero(float* p, long n, hipStream_t s) {
  fill_u32_kernel<<<blks(n), 256, 0, s>>>((unsigned*)p, 0u, n);
}

static void run_tconv(const TcP& p, const float* x, int Din, int n,
                      const int* src, const int* dst, int E, const float* ea,
                      float* out, Scratch& w, hipStream_t s) {
  // fused Q|K|V|Skip projection: one A read, N = 1024
  conv_w(p.Wq, w.WBF + (size_t)0   * Din, Din, 256, s);
  conv_w(p.Wk, w.WBF + (size_t)256 * Din, Din, 256, s);
  conv_w(p.Wv, w.WBF + (size_t)512 * Din, Din, 256, s);
  conv_w(p.Ws, w.WBF + (size_t)768 * Din, Din, 256, s);
  pack_bias_kernel<<<blks(1024), 256, 0, s>>>(w.BIASP, p.bq, p.bk, p.bv, p.bs, 256, 4);
  gemm(x, w.WBF, w.BIASP, w.FQKVS, n, 1024, Din, 0, 1024, s);
  // edge projection
  conv_w(p.We, w.WBF, 64, 256, s);
  gemm(ea, w.WBF, nullptr, w.FEP, E, 256, 64, 0, 256, s);
  // segment softmax + aggregate
  fill_u32_kernel<<<blks((long)n * 4), 256, 0, s>>>(w.UM, 0u, (long)n * 4);
  zero(w.FDEN, (long)n * 4, s);
  zero(out, (long)n * 256, s);
  tc_score_kernel<<<blks((long)E * 4), 256, 0, s>>>(w.FQKVS, w.FQKVS + 256, w.FEP,
                                                    src, dst, w.FES, w.UM, E, 1024);
  seg_exp_kernel<<<blks((long)E * 4), 256, 0, s>>>(w.FES, w.UM, dst, w.FEX, w.FDEN, E, 4);
  tc_scatter_kernel<<<blks((long)E * 256), 256, 0, s>>>(w.FQKVS + 512, 1024, w.FEP,
                                                        w.FEX, w.FDEN, src, dst, out, E);
  vadd_strided_kernel<<<blks((long)n * 256), 256, 0, s>>>(out, w.FQKVS + 768, 1024, (long)n * 256);
}

static void run_seq(const SeqP& P, const float* xin, int n,
                    const int* src, const int* dst, int E, const float* ea,
                    float* out, Scratch& w, hipStream_t s) {
  const int D = 64;
  // --- GINEConv ---
  conv_w(P.g.We, w.WBF, D, D, s);
  gemm(ea, w.WBF, P.g.be, w.FEG, E, D, D, 0, D, s);
  zero(w.FAGG, (long)n * D, s);
  gine_msg_kernel<<<blks((long)E * D), 256, 0, s>>>(xin, w.FEG, src, dst, w.FAGG, E);
  gine_comb_kernel<<<blks((long)n * D), 256, 0, s>>>(xin, w.FAGG, P.g.eps, w.FA, (long)n * D);
  conv_w(P.g.W1, w.WBF, D, D, s); gemm(w.FA, w.WBF, P.g.b1, w.FB, n, D, D, 1, D, s);
  conv_w(P.g.W2, w.WBF, D, D, s); gemm(w.FB, w.WBF, P.g.b2, w.FA, n, D, D, 0, D, s);
  // --- TransformerConv 1 (64 -> 256), x: FA -> FB ---
  run_tconv(P.t1, w.FA, D, n, src, dst, E, ea, w.FB, w, s);
  // --- TransformerConv 2 (256 -> 256), x: FB -> FA ---
  run_tconv(P.t2, w.FB, 256, n, src, dst, E, ea, w.FA, w, s);
  // --- GATv2 (256 -> 64): fused xl|xr projection, N = 128 ---
  conv_w(P.ga.Wl, w.WBF, 256, D, s);
  conv_w(P.ga.Wr, w.WBF + (size_t)D * 256, 256, D, s);
  pack_bias_kernel<<<blks(128), 256, 0, s>>>(w.BIASP, P.ga.bl, P.ga.br, P.ga.bl, P.ga.bl, D, 2);
  gemm(w.FA, w.WBF, w.BIASP, w.FQKVS, n, 128, 256, 0, 128, s);
  conv_w(P.ga.We, w.WBF, D, D, s);
  gemm(ea, w.WBF, nullptr, w.FEG, E, D, D, 0, D, s);
  fill_u32_kernel<<<blks(n), 256, 0, s>>>(w.UM, 0u, n);
  zero(w.FDEN, n, s);
  zero(out, (long)n * D, s);
  gat_score_kernel<<<blks(E), 256, 0, s>>>(w.FQKVS, w.FQKVS + 64, 128, w.FEG, P.ga.att,
                                           src, dst, w.FES, w.UM, E);
  seg_exp_kernel<<<blks(E), 256, 0, s>>>(w.FES, w.UM, dst, w.FEX, w.FDEN, E, 1);
  gat_scatter_kernel<<<blks((long)E * D), 256, 0, s>>>(w.FQKVS, 128, w.FEX, w.FDEN,
                                                       src, dst, out, E);
  add_bias_kernel<<<blks((long)n * D), 256, 0, s>>>(out, P.ga.bo, (long)n * D);
}

extern "C" void kernel_launch(void* const* d_in, const int* in_sizes, int n_in,
                              void* d_out, int out_size, void* d_ws, size_t ws_size,
                              hipStream_t stream) {
  // ---- inputs (setup_inputs dict insertion order) ----
  const float* x          = (const float*)d_in[0];
  const int*   edge_index = (const int*)  d_in[1];
  const float* edge_attr  = (const float*)d_in[2];
  const int*   line_ei    = (const int*)  d_in[3];
  const int*   line_sh    = (const int*)  d_in[4];

  const int Nn = in_sizes[0] / 64;       // 50000
  const int E  = in_sizes[1] / 2;        // 200000
  const int NL = in_sizes[2] / 64;       // 200000 (line-graph node count)
  const int EL = in_sizes[3] / 2;        // 200000 (line-graph edge count)

  // ---- params: nested dicts flattened in insertion order (96 leaves) ----
  int idx = 5;
  auto nf = [&]() { return (const float*)d_in[idx++]; };
  SeqP sp[3];
  for (int si = 0; si < 3; ++si) {
    SeqP& P = sp[si];
    P.g.We = nf(); P.g.be = nf(); P.g.W1 = nf(); P.g.b1 = nf();
    P.g.W2 = nf(); P.g.b2 = nf(); P.g.eps = nf();
    TcP* ts[2] = { &P.t1, &P.t2 };
    for (int ti = 0; ti < 2; ++ti) {
      TcP& T = *ts[ti];
      T.Wq = nf(); T.bq = nf(); T.Wk = nf(); T.bk = nf();
      T.Wv = nf(); T.bv = nf(); T.We = nf(); T.Ws = nf(); T.bs = nf();
    }
    P.ga.Wl = nf(); P.ga.bl = nf(); P.ga.Wr = nf(); P.ga.br = nf();
    P.ga.We = nf(); P.ga.att = nf(); P.ga.bo = nf();
  }
  const float* wts = nf();               // weights[3]

  // ---- workspace carve ----
  const long nmax = (Nn > NL) ? Nn : NL;
  const long Emax = (E > EL) ? E : EL;
  uint8_t* cur = (uint8_t*)d_ws;
  auto carve = [&](size_t bytes) -> void* {
    void* p = cur; cur += (bytes + 255) & ~(size_t)255; return p;
  };
  Scratch w;
  w.FA    = (float*)carve((size_t)nmax * 256 * 4);
  w.FB    = (float*)carve((size_t)nmax * 256 * 4);
  w.FQKVS = (float*)carve((size_t)nmax * 1024 * 4);    // fused q|k|v|skip (or xl|xr)
  w.FEP   = (float*)carve((size_t)Emax * 256 * 4);
  w.FEG   = (float*)carve((size_t)Emax * 64 * 4);
  w.FAGG  = (float*)carve((size_t)nmax * 64 * 4);
  w.FES   = (float*)carve((size_t)Emax * 4 * 4);
  w.FEX   = (float*)carve((size_t)Emax * 4 * 4);
  w.FDEN  = (float*)carve((size_t)nmax * 4 * 4);
  w.BIASP = (float*)carve((size_t)1024 * 4);
  w.UM    = (unsigned*)carve((size_t)nmax * 4 * 4);
  w.WBF   = (unsigned short*)carve((size_t)1024 * 256 * 2);
  float* FLEA = (float*)carve((size_t)EL * 64 * 4);
  float* FX2  = (float*)carve((size_t)Nn * 64 * 4);
  float* FEA1 = (float*)carve((size_t)NL * 64 * 4);
  float* FEA3 = (float*)carve((size_t)NL * 64 * 4);
  if ((size_t)(cur - (uint8_t*)d_ws) > ws_size) return;   // not enough scratch

  const int* src  = edge_index;
  const int* dst  = edge_index + E;
  const int* lsrc = line_ei;
  const int* ldst = line_ei + EL;

  // seq2 on the original graph: x2
  run_seq(sp[1], x, Nn, src, dst, E, edge_attr, FX2, w, stream);
  // line-graph edge features: lea = x[line_shared]
  gather_rows_kernel<<<blks((long)EL * 64), 256, 0, stream>>>(x, line_sh, FLEA, EL);
  // seq1 / seq3 on the line graph (nodes = original edges)
  run_seq(sp[0], edge_attr, NL, lsrc, ldst, EL, FLEA, FEA1, w, stream);
  run_seq(sp[2], edge_attr, NL, lsrc, ldst, EL, FLEA, FEA3, w, stream);

  // final weighted combine + edge_index pass-through
  float* out_x  = (float*)d_out;
  float* out_ea = out_x + (size_t)Nn * 64;
  int*   out_ei = (int*)(out_ea + (size_t)NL * 64);
  combine_x_kernel<<<blks((long)Nn * 64), 256, 0, stream>>>(x, FX2, wts, out_x, (long)Nn * 64);
  combine_ea_kernel<<<blks((long)NL * 64), 256, 0, stream>>>(FEA1, edge_attr, FEA3, wts, out_ea, (long)NL * 64);
  copy_i32_kernel<<<blks((long)2 * E), 256, 0, stream>>>(edge_index, out_ei, (long)2 * E);
}
